// HyenaFilter_31868657336411
// MI455X (gfx1250) — matrix operand order repaired and checked by reading, not compiled, not run on hardware
//
#include <hip/hip_runtime.h>

// Problem constants (match the reference)
#define B_    4
#define C_    768
#define L_    8192
#define E_    33
#define O_    64
#define KRLEN (L_ + 32)   // reversed filter with 16-halfword zero pad on each side

typedef __attribute__((ext_vector_type(16))) _Float16 v16h;
typedef __attribute__((ext_vector_type(8)))  _Float16 v8h;
typedef __attribute__((ext_vector_type(8)))  float    v8f;

// ---------------------------------------------------------------------------
// Kernel 1: implicit filter MLP ([Dense->Sin]x3 -> Dense) + exp modulation.
// Writes reversed, zero-padded f16 filter:  kr[c, a] = k[c, L+15-a]
// ---------------------------------------------------------------------------
__global__ __launch_bounds__(256) void hyena_filter_kernel(
    const float* __restrict__ z,      // [L, E]
    const float* __restrict__ t,      // [L]
    const float* __restrict__ deltas, // [C]
    const float* __restrict__ freq,   // [O]
    const float* __restrict__ w0, const float* __restrict__ b0,   // [E,O],[O]
    const float* __restrict__ w1, const float* __restrict__ b1,   // [O,O],[O]
    const float* __restrict__ w2, const float* __restrict__ b2,   // [O,O],[O]
    const float* __restrict__ wout,                               // [O,C]
    _Float16* __restrict__ kr)                                    // [C, KRLEN]
{
  __shared__ float ws0[E_ * O_];
  __shared__ float ws1[O_ * O_];
  __shared__ float ws2[O_ * O_];
  __shared__ float sb0[O_], sb1[O_], sb2[O_], sfreq[O_];
  __shared__ float zl[16 * E_];
  __shared__ float hA[16 * O_];
  __shared__ float hB[16 * O_];

  const int tid = threadIdx.x;
  const int blk = blockIdx.x;
  const int l0  = blk * 16;

  for (int idx = tid; idx < E_ * O_; idx += 256) ws0[idx] = w0[idx];
  for (int idx = tid; idx < O_ * O_; idx += 256) ws1[idx] = w1[idx];
  for (int idx = tid; idx < O_ * O_; idx += 256) ws2[idx] = w2[idx];
  if (tid < O_) { sb0[tid] = b0[tid]; sb1[tid] = b1[tid]; sb2[tid] = b2[tid]; sfreq[tid] = freq[tid]; }
  for (int idx = tid; idx < 16 * E_; idx += 256) zl[idx] = z[l0 * E_ + idx];
  __syncthreads();

  #pragma unroll
  for (int q = 0; q < 4; ++q) {
    int idx = q * 256 + tid;
    int i = idx >> 6, o = idx & 63;
    float acc = sb0[o];
    for (int e = 0; e < E_; ++e) acc += zl[i * E_ + e] * ws0[e * O_ + o];
    hA[i * O_ + o] = __sinf(sfreq[o] * acc);
  }
  __syncthreads();

  #pragma unroll
  for (int q = 0; q < 4; ++q) {
    int idx = q * 256 + tid;
    int i = idx >> 6, o = idx & 63;
    float acc = sb1[o];
    for (int e = 0; e < O_; ++e) acc += hA[i * O_ + e] * ws1[e * O_ + o];
    hB[i * O_ + o] = __sinf(sfreq[o] * acc);
  }
  __syncthreads();

  #pragma unroll
  for (int q = 0; q < 4; ++q) {
    int idx = q * 256 + tid;
    int i = idx >> 6, o = idx & 63;
    float acc = sb2[o];
    for (int e = 0; e < O_; ++e) acc += hB[i * O_ + e] * ws2[e * O_ + o];
    hA[i * O_ + o] = __sinf(sfreq[o] * acc);
  }
  __syncthreads();

  for (int q = 0; q < 48; ++q) {
    int idx = q * 256 + tid;
    int i = idx / C_, c = idx % C_;
    float acc = 0.f;
    for (int o = 0; o < O_; ++o) acc += hA[i * O_ + o] * wout[o * C_ + c];
    int l = l0 + i;
    float m = __expf(-t[l] * fabsf(deltas[c]));
    kr[(size_t)c * KRLEN + (L_ + 15 - l)] = (_Float16)(acc * m);
  }

  if (tid < 32) {
    int a = (tid < 16) ? tid : (L_ + tid);
    kr[(size_t)blk * KRLEN + a] = (_Float16)0.f;
    int c2 = blk + 512;
    if (c2 < C_) kr[(size_t)c2 * KRLEN + a] = (_Float16)0.f;
  }
}

// ---------------------------------------------------------------------------
// Kernel 1.5: optional x f32 -> f16 pre-conversion (fast A-operand path).
// ---------------------------------------------------------------------------
__global__ __launch_bounds__(256) void hyena_cvt_x_kernel(
    const float* __restrict__ x, _Float16* __restrict__ xh)
{
  size_t idx = (size_t)blockIdx.x * 256 + threadIdx.x;   // chunk of 8
  const float4* q = (const float4*)(x + idx * 8);
  float4 u = q[0], v = q[1];
  v8h r;
  r[0] = (_Float16)u.x; r[1] = (_Float16)u.y; r[2] = (_Float16)u.z; r[3] = (_Float16)u.w;
  r[4] = (_Float16)v.x; r[5] = (_Float16)v.y; r[6] = (_Float16)v.z; r[7] = (_Float16)v.w;
  *(v8h*)(xh + idx * 8) = r;
}

// -------------------- A-operand loaders (8 elems, 32B/16B aligned) ---------
__device__ __forceinline__ v8h ld8(const float* __restrict__ xr, int p) {
  const float4* q = (const float4*)(xr + p);
  float4 u = q[0], v = q[1];
  v8h r;
  r[0] = (_Float16)u.x; r[1] = (_Float16)u.y; r[2] = (_Float16)u.z; r[3] = (_Float16)u.w;
  r[4] = (_Float16)v.x; r[5] = (_Float16)v.y; r[6] = (_Float16)v.z; r[7] = (_Float16)v.w;
  return r;
}
__device__ __forceinline__ v8h ld8(const _Float16* __restrict__ xr, int p) {
  return *(const v8h*)(xr + p);
}
template <typename XT>
__device__ __forceinline__ v8h ld8z(const XT* __restrict__ xr, int p) {
  v8h r = {};
  if (p >= 0) r = ld8(xr, p);
  return r;
}
template <typename XT>
__device__ __forceinline__ v16h ldA(const XT* __restrict__ xr, int p) {
  v8h lo = ld8(xr, p), hi = ld8(xr, p + 8);
  return __builtin_shufflevector(lo, hi, 0,1,2,3,4,5,6,7,8,9,10,11,12,13,14,15);
}
template <typename XT>
__device__ __forceinline__ v16h ldAz(const XT* __restrict__ xr, int p) {
  v8h lo = ld8z(xr, p), hi = ld8z(xr, p + 8);
  return __builtin_shufflevector(lo, hi, 0,1,2,3,4,5,6,7,8,9,10,11,12,13,14,15);
}

// ---------------------------------------------------------------------------
// Kernel 2: causal long convolution via block-Toeplitz WMMA (16x16x32 f16).
//  M = block index I, N = within-block position i, K = two diagonals {D,D+1}.
//  A (lane row I): two aligned x chunks; B (lane col i): 16 contiguous halfs
//  of the reversed filter staged in LDS via async global->LDS copies.
//  One wave per (c, group of 16 output blocks); 4 accumulators = 4 batches.
// ---------------------------------------------------------------------------
template <typename XT>
__global__ __launch_bounds__(256) void hyena_conv_kernel(
    const float*    __restrict__ x32,   // [B, C, L] f32 (epilogue bias-add)
    const XT*       __restrict__ xa,    // [B, C, L] A-operand source
    const float*    __restrict__ bias,  // [C]
    const _Float16* __restrict__ kr,    // [C, KRLEN]
    float*          __restrict__ out)   // [B, C, L]
{
  __shared__ _Float16 krs[KRLEN];

  const int tid = threadIdx.x;
  const int c   = blockIdx.x >> 2;
  const int gq  = blockIdx.x & 3;

  // Stage this channel's reversed filter into LDS with async copies
  // (ASYNCcnt-tracked global->LDS DMA path), then fence + barrier.
  {
    const _Float16* src = kr + (size_t)c * KRLEN;
    for (int idx = tid; idx < KRLEN / 8; idx += 256) {   // 1028 16B chunks
      unsigned lds_off = (unsigned)(uintptr_t)(&krs[idx * 8]);
      unsigned long long gaddr = (unsigned long long)(uintptr_t)(src + idx * 8);
      asm volatile("global_load_async_to_lds_b128 %0, %1, off"
                   :: "v"(lds_off), "v"(gaddr) : "memory");
    }
    asm volatile("s_wait_asynccnt 0x0" ::: "memory");
  }
  __syncthreads();

  const int w     = tid >> 5;
  const int lane  = tid & 31;
  const int i_n   = lane & 15;
  const int h     = lane >> 4;
  const int g     = w * 4 + gq;      // 0..31
  const int I0    = g * 16;
  const int nmain = 8 * g;           // unguarded diagonal pairs
  const int niter = 8 * g + 8;       // total diagonal pairs

  const XT* xr0 = xa + (size_t)(0 * C_ + c) * L_;
  const XT* xr1 = xa + (size_t)(1 * C_ + c) * L_;
  const XT* xr2 = xa + (size_t)(2 * C_ + c) * L_;
  const XT* xr3 = xa + (size_t)(3 * C_ + c) * L_;

  v8f acc0 = {}, acc1 = {}, acc2 = {}, acc3 = {};

  int aB = (L_ - 1 - i_n) + 16 * h;          // halfword index into krs
  int p0 = 16 * (I0 + i_n - 1) + 16 * h;     // element index into x row

  // Main loop: all lanes in-bounds -> pure loads + 4 WMMAs, no branches.
  for (int it = 0; it < nmain; ++it) {
    v16h bv;
    #pragma unroll
    for (int e = 0; e < 16; ++e) bv[e] = krs[aB + e];

    v16h a0 = ldA(xr0, p0);
    v16h a1 = ldA(xr1, p0);
    v16h a2 = ldA(xr2, p0);
    v16h a3 = ldA(xr3, p0);

    acc0 = __builtin_amdgcn_wmma_f32_16x16x32_f16(false, a0, false, bv, (short)0, acc0, false, false);
    acc1 = __builtin_amdgcn_wmma_f32_16x16x32_f16(false, a1, false, bv, (short)0, acc1, false, false);
    acc2 = __builtin_amdgcn_wmma_f32_16x16x32_f16(false, a2, false, bv, (short)0, acc2, false, false);
    acc3 = __builtin_amdgcn_wmma_f32_16x16x32_f16(false, a3, false, bv, (short)0, acc3, false, false);

    aB -= 32;
    p0 -= 32;
  }

  // Tail (8 iterations): causal left edge -> zero out-of-range chunks.
  for (int it = nmain; it < niter; ++it) {
    v16h bv;
    #pragma unroll
    for (int e = 0; e < 16; ++e) bv[e] = krs[aB + e];

    v16h a0 = ldAz(xr0, p0);
    v16h a1 = ldAz(xr1, p0);
    v16h a2 = ldAz(xr2, p0);
    v16h a3 = ldAz(xr3, p0);

    acc0 = __builtin_amdgcn_wmma_f32_16x16x32_f16(false, a0, false, bv, (short)0, acc0, false, false);
    acc1 = __builtin_amdgcn_wmma_f32_16x16x32_f16(false, a1, false, bv, (short)0, acc1, false, false);
    acc2 = __builtin_amdgcn_wmma_f32_16x16x32_f16(false, a2, false, bv, (short)0, acc2, false, false);
    acc3 = __builtin_amdgcn_wmma_f32_16x16x32_f16(false, a3, false, bv, (short)0, acc3, false, false);

    aB -= 32;
    p0 -= 32;
  }

  // Epilogue: lane holds (N=i_n, M=r+8h) => l = 16*(I0+r+8h)+i_n
  const float bc = bias[c];
  #pragma unroll
  for (int r = 0; r < 8; ++r) {
    const size_t l = (size_t)16 * (I0 + r + 8 * h) + i_n;
    size_t o0 = (size_t)(0 * C_ + c) * L_ + l;
    size_t o1 = (size_t)(1 * C_ + c) * L_ + l;
    size_t o2 = (size_t)(2 * C_ + c) * L_ + l;
    size_t o3 = (size_t)(3 * C_ + c) * L_ + l;
    out[o0] = acc0[r] + x32[o0] * bc;
    out[o1] = acc1[r] + x32[o1] * bc;
    out[o2] = acc2[r] + x32[o2] * bc;
    out[o3] = acc3[r] + x32[o3] * bc;
  }
}

// ---------------------------------------------------------------------------
extern "C" void kernel_launch(void* const* d_in, const int* in_sizes, int n_in,
                              void* d_out, int out_size, void* d_ws, size_t ws_size,
                              hipStream_t stream) {
  const float* x      = (const float*)d_in[0];
  const float* bias   = (const float*)d_in[1];
  const float* z      = (const float*)d_in[2];
  const float* t      = (const float*)d_in[3];
  const float* deltas = (const float*)d_in[4];
  const float* freq   = (const float*)d_in[5];
  const float* w0     = (const float*)d_in[6];
  const float* b0     = (const float*)d_in[7];
  const float* w1     = (const float*)d_in[8];
  const float* b1     = (const float*)d_in[9];
  const float* w2     = (const float*)d_in[10];
  const float* b2     = (const float*)d_in[11];
  const float* wout   = (const float*)d_in[12];

  const size_t kr_bytes = (size_t)C_ * KRLEN * sizeof(_Float16);   // ~12.6 MB
  const size_t xh_bytes = (size_t)B_ * C_ * L_ * sizeof(_Float16); // ~50.3 MB

  _Float16* kr  = (_Float16*)d_ws;
  float*    out = (float*)d_out;

  hyena_filter_kernel<<<L_ / 16, 256, 0, stream>>>(
      z, t, deltas, freq, w0, b0, w1, b1, w2, b2, wout, kr);

  if (ws_size >= kr_bytes + xh_bytes) {
    // Fast path: pre-convert x to f16 -> halves A-operand bytes, no cvts in loop.
    _Float16* xh = (_Float16*)((char*)d_ws + kr_bytes);
    hyena_cvt_x_kernel<<<(B_ * C_ * L_) / (8 * 256), 256, 0, stream>>>(x, xh);
    hyena_conv_kernel<_Float16><<<C_ * 4, 256, 0, stream>>>(x, xh, bias, kr, out);
  } else {
    hyena_conv_kernel<float><<<C_ * 4, 256, 0, stream>>>(x, x, bias, kr, out);
  }
}
